// GroundTruthiDBFDynamics_31379031064759
// MI455X (gfx1250) — compile-verified, matrix operand order checked by reading, not run on hardware
//
#include <hip/hip_runtime.h>
#include <math.h>

typedef __attribute__((ext_vector_type(16))) __bf16 v16bf;
typedef __attribute__((ext_vector_type(8)))  __bf16 v8bf;
typedef __attribute__((ext_vector_type(8)))  float  v8f;

#define LATENT 256
#define HDIM   64
#define NCTRL  2
#define BB     256
#define TT     200
#define DD     32
#define WIN    5
#define DTSTEP 0.1f
#define NSUB   2

// LDS row pitches (bf16 elems) chosen so pitch in DWORDs == 4*odd (mod 64):
// 16 fragment rows then tile all 64 banks in 4-bank chunks -> conflict-free
// ds_load_b128. 64->72 (36 DW), 128->136 (68 DW), 256->264 (132 DW).
#define P64   72
#define P128  136
#define P256  264

// ---- WMMA helpers (CDNA5 16x16x32 bf16, f32 accum) --------------------------
// A (16-bit 16x32): lane l -> row = l&15; vec[0..7] = K k0+8*(l>>4)+j,
// vec[8..15] = K k0+16+8*(l>>4)+j  => two contiguous 16B LDS loads.
// B (32x16) mirrors A with N in place of M (weights stored transposed [out][in]).
__device__ __forceinline__ v16bf load_frag(const __bf16* rowp, int k0, int hi) {
  v8bf lo = *(const v8bf*)(rowp + k0 + 8 * hi);
  v8bf hh = *(const v8bf*)(rowp + k0 + 16 + 8 * hi);
  v16bf a;
#pragma unroll
  for (int j = 0; j < 8; ++j) { a[j] = lo[j]; a[j + 8] = hh[j]; }
  return a;
}

__device__ __forceinline__ v8f wmma_bf16(v16bf a, v16bf b, v8f c) {
  return __builtin_amdgcn_wmma_f32_16x16x32_bf16(false, a, false, b, (short)0, c,
                                                 false, false);
}

// Branchless tanh: (e^2x - 1)/(e^2x + 1), raw v_exp_f32 + v_rcp_f32.
// Input clamped so exp2 arg stays in [-23.1, 23.1]; accuracy ~1e-7, far
// beyond bf16 operand precision. ~6 VALU, no exec-mask divergence.
__device__ __forceinline__ float tanh_fast(float x) {
  float xc = fminf(8.f, fmaxf(-8.f, x));
  float e  = __builtin_amdgcn_exp2f(xc * 2.88539008178f);  // 2*log2(e)
  return (e - 1.f) * __builtin_amdgcn_rcpf(e + 1.f);
}

// ============================================================================
// Encoder: Deep-Sets phi(1->64->64) sum_D, rho(64->256). One WG per
// (batch b, 16-timestep chunk). 512 phi-rows per WG, WMMA for phi2 and rho.
// ============================================================================
__global__ void __launch_bounds__(256) encoder_kernel(
    const float* __restrict__ gi,
    const float* __restrict__ phi_w1, const float* __restrict__ phi_b1,
    const float* __restrict__ phi_w2, const float* __restrict__ phi_b2,
    const float* __restrict__ rho_w,  const float* __restrict__ rho_b,
    float* __restrict__ xs_out)
{
  extern __shared__ unsigned char smem[];
  float*  s_phiw1 = (float*)smem;                  // 64
  float*  s_phib1 = s_phiw1 + 64;                  // 64
  float*  s_phib2 = s_phib1 + 64;                  // 64
  float*  s_rhob  = s_phib2 + 64;                  // 256
  __bf16* s_h1    = (__bf16*)(s_rhob + 256);       // [512][P64]
  __bf16* s_h2    = s_h1 + 512 * P64;              // [512][P64]
  __bf16* s_w2t   = s_h2 + 512 * P64;              // [64][P64]  (phi_w2^T)
  __bf16* s_rhot  = s_w2t + 64 * P64;              // [256][P64] (rho_w^T)
  __bf16* s_S     = s_rhot + 256 * P64;            // [16][P64]

  const int tid = threadIdx.x;
  const int lane = tid & 31, wave = tid >> 5;
  const int hi = lane >> 4, lm = lane & 15;
  const int b  = blockIdx.y;
  const int t0 = blockIdx.x * 16;

  for (int idx = tid; idx < 64; idx += 256) {
    s_phiw1[idx] = phi_w1[idx]; s_phib1[idx] = phi_b1[idx]; s_phib2[idx] = phi_b2[idx];
  }
  for (int idx = tid; idx < 256; idx += 256) s_rhob[idx] = rho_b[idx];
  for (int idx = tid; idx < 64 * 64; idx += 256) {
    int i = idx >> 6, o = idx & 63;
    s_w2t[o * P64 + i] = (__bf16)phi_w2[idx];
  }
  for (int idx = tid; idx < 64 * 256; idx += 256) {
    int i = idx >> 8, o = idx & 255;
    s_rhot[o * P64 + i] = (__bf16)rho_w[idx];
  }

  // phi1 (rank-1): rows r = tl*32 + d
  for (int r = tid; r < 512; r += 256) {
    int tl = r >> 5, d = r & 31;
    int t = t0 + tl;
    float e = (t < TT) ? gi[((size_t)b * TT + t) * DD + d] : 0.f;
#pragma unroll 8
    for (int j = 0; j < 64; ++j)
      s_h1[r * P64 + j] = (__bf16)tanh_fast(e * s_phiw1[j] + s_phib1[j]);
  }
  __syncthreads();

  // phi2: [512,64] x [64,64], 32 M-tiles x 4 N-tiles, K=64
  for (int job = wave; job < 128; job += 8) {
    int mt = job >> 2, nt = job & 3;
    v8f c = {};
#pragma unroll
    for (int k0 = 0; k0 < 64; k0 += 32) {
      v16bf a  = load_frag(s_h1  + (mt * 16 + lm) * P64, k0, hi);
      v16bf bm = load_frag(s_w2t + (nt * 16 + lm) * P64, k0, hi);
      c = wmma_bf16(a, bm, c);
    }
    int n = nt * 16 + lm;
#pragma unroll
    for (int v = 0; v < 8; ++v)
      s_h2[(mt * 16 + v + 8 * hi) * P64 + n] = (__bf16)tanh_fast(c[v] + s_phib2[n]);
  }
  __syncthreads();

  // sum over the 32 quantities per timestep
  for (int idx = tid; idx < 16 * 64; idx += 256) {
    int tl = idx >> 6, j = idx & 63;
    float s = 0.f;
#pragma unroll 8
    for (int d = 0; d < 32; ++d) s += (float)s_h2[(tl * 32 + d) * P64 + j];
    s_S[tl * P64 + j] = (__bf16)s;
  }
  __syncthreads();

  // rho: [16,64] x [64,256], 16 N-tiles, K=64
  for (int nt = wave; nt < 16; nt += 8) {
    v8f c = {};
#pragma unroll
    for (int k0 = 0; k0 < 64; k0 += 32) {
      v16bf a  = load_frag(s_S    + lm * P64,              k0, hi);
      v16bf bm = load_frag(s_rhot + (nt * 16 + lm) * P64,  k0, hi);
      c = wmma_bf16(a, bm, c);
    }
    int n = nt * 16 + lm;
#pragma unroll
    for (int v = 0; v < 8; ++v) {
      int t = t0 + v + 8 * hi;
      if (t < TT)
        xs_out[((size_t)b * TT + t) * LATENT + n] = c[v] + s_rhob[n];
    }
  }
}

// ============================================================================
// Scan: one WG (512 threads / 16 waves) per 16 batch rows; all f/g weights
// resident in LDS (bf16, transposed, bank-conflict-free pitches).
// 199 steps x 2 RK4 substeps x 4 stages; each stage = 3 LDS WMMA layers +
// elementwise combine; only resets/u/outputs touch HBM.
// ============================================================================
__global__ void __launch_bounds__(512) scan_kernel(
    const float* __restrict__ gu,
    const float* __restrict__ f_w1, const float* __restrict__ f_b1,
    const float* __restrict__ f_w2, const float* __restrict__ f_b2,
    const float* __restrict__ f_w3, const float* __restrict__ f_b3,
    const float* __restrict__ g_w1, const float* __restrict__ g_b1,
    const float* __restrict__ g_w2, const float* __restrict__ g_b2,
    const float* __restrict__ g_w3, const float* __restrict__ g_b3,
    const float* __restrict__ xs,   float* __restrict__ tides)
{
  extern __shared__ unsigned char smem[];
  float*  s_x0   = (float*)smem;         // [16][256] RK4 substep base / carry
  float*  s_acc  = s_x0  + 4096;         // [16][256] k accumulator
  float*  s_fout = s_acc + 4096;         // [16][256]
  float*  s_gout = s_fout + 4096;        // [16][512]
  float*  s_fb1  = s_gout + 8192;        // 64
  float*  s_gb1  = s_fb1 + 64;
  float*  s_fb2  = s_gb1 + 64;
  float*  s_gb2  = s_fb2 + 64;
  float*  s_fb3  = s_gb2 + 64;           // 256
  float*  s_gb3  = s_fb3 + 256;          // 512
  float*  s_uk   = s_gb3 + 512;          // [16][2]
  __bf16* s_w1   = (__bf16*)(s_uk + 32); // [128][P256] = [f_w1^T ; g_w1^T]
  __bf16* s_fw2  = s_w1  + 128 * P256;   // [64][P64]
  __bf16* s_gw2  = s_fw2 + 64 * P64;     // [64][P64]
  __bf16* s_fw3  = s_gw2 + 64 * P64;     // [256][P64]
  __bf16* s_gw3  = s_fw3 + 256 * P64;    // [512][P64]
  __bf16* s_xbf  = s_gw3 + 512 * P64;    // [16][P256] eval point (A matrix)
  __bf16* s_h1   = s_xbf + 16 * P256;    // [16][P128] = [h1f | h1g]
  __bf16* s_h2   = s_h1  + 16 * P128;    // [16][P128] = [h2f | h2g]

  const int tid = threadIdx.x;
  const int lane = tid & 31, wave = tid >> 5;
  const int hi = lane >> 4, lm = lane & 15;
  const int b0 = blockIdx.x * 16;

  // ---- load+transpose weights into LDS (once) ----
  for (int idx = tid; idx < 256 * 64; idx += 512) {
    int i = idx >> 6, o = idx & 63;
    s_w1[o * P256 + i]        = (__bf16)f_w1[idx];
    s_w1[(64 + o) * P256 + i] = (__bf16)g_w1[idx];
  }
  for (int idx = tid; idx < 64 * 64; idx += 512) {
    int i = idx >> 6, o = idx & 63;
    s_fw2[o * P64 + i] = (__bf16)f_w2[idx];
    s_gw2[o * P64 + i] = (__bf16)g_w2[idx];
  }
  for (int idx = tid; idx < 64 * 256; idx += 512) {
    int i = idx >> 8, o = idx & 255;
    s_fw3[o * P64 + i] = (__bf16)f_w3[idx];
  }
  for (int idx = tid; idx < 64 * 512; idx += 512) {
    int i = idx >> 9, o = idx & 511;
    s_gw3[o * P64 + i] = (__bf16)g_w3[idx];
  }
  for (int idx = tid; idx < 64; idx += 512) {
    s_fb1[idx] = f_b1[idx]; s_gb1[idx] = g_b1[idx];
    s_fb2[idx] = f_b2[idx]; s_gb2[idx] = g_b2[idx];
  }
  for (int idx = tid; idx < 256; idx += 512) s_fb3[idx] = f_b3[idx];
  for (int idx = tid; idx < 512; idx += 512) s_gb3[idx] = g_b3[idx];

  // tides[:,0] = xs[:,0]
  for (int q = 0; q < 8; ++q) {
    int e = tid * 8 + q; int r = e >> 8, l = e & 255;
    tides[((size_t)(b0 + r) * TT) * LATENT + l] =
        xs[((size_t)(b0 + r) * TT) * LATENT + l];
  }
  __syncthreads();

  const float hh = DTSTEP / NSUB;

  for (int k = 1; k < TT; ++k) {
    if ((k % WIN) == 1) {  // window reset: carry = xs[:, k-1]
      for (int q = 0; q < 8; ++q) {
        int e = tid * 8 + q; int r = e >> 8, l = e & 255;
        s_x0[e] = xs[((size_t)(b0 + r) * TT + (k - 1)) * LATENT + l];
      }
    }
    if (tid < 32) {  // u[:, k]
      int r = tid >> 1, c = tid & 1;
      s_uk[tid] = gu[((size_t)(b0 + r) * TT + k) * NCTRL + c];
    }

    for (int s = 0; s < NSUB; ++s) {
      __syncthreads();
      for (int q = 0; q < 8; ++q) {
        int e = tid * 8 + q; int r = e >> 8, l = e & 255;
        s_acc[e] = 0.f;
        s_xbf[r * P256 + l] = (__bf16)s_x0[e];
      }
      __syncthreads();

      for (int st = 0; st < 4; ++st) {
        // ---- stage 1: x[16,256] x [w1f|w1g]^T -> h1[16,128] (waves 0-7) ----
        if (wave < 8) {
          v8f c = {};
#pragma unroll
          for (int k0 = 0; k0 < 256; k0 += 32) {
            v16bf a  = load_frag(s_xbf + lm * P256,                k0, hi);
            v16bf bm = load_frag(s_w1  + (wave * 16 + lm) * P256,  k0, hi);
            c = wmma_bf16(a, bm, c);
          }
          int n = wave * 16 + lm;
          float bias = (n < 64) ? s_fb1[n] : s_gb1[n - 64];
#pragma unroll
          for (int v = 0; v < 8; ++v)
            s_h1[(v + 8 * hi) * P128 + n] = (__bf16)tanh_fast(c[v] + bias);
        }
        __syncthreads();
        // ---- stage 2: h1f x f_w2 (waves 0-3), h1g x g_w2 (waves 4-7) ----
        if (wave < 8) {
          int isg = wave >> 2, nt = wave & 3;
          const __bf16* w2p = isg ? s_gw2 : s_fw2;
          v8f c = {};
#pragma unroll
          for (int k0 = 0; k0 < 64; k0 += 32) {
            v16bf a  = load_frag(s_h1 + lm * P128 + isg * 64,  k0, hi);
            v16bf bm = load_frag(w2p + (nt * 16 + lm) * P64,   k0, hi);
            c = wmma_bf16(a, bm, c);
          }
          int n = nt * 16 + lm;
          float bias = isg ? s_gb2[n] : s_fb2[n];
#pragma unroll
          for (int v = 0; v < 8; ++v)
            s_h2[(v + 8 * hi) * P128 + isg * 64 + n] = (__bf16)tanh_fast(c[v] + bias);
        }
        __syncthreads();
        // ---- stage 3: h2f x f_w3 (16 tiles) + h2g x g_w3 (32 tiles), 16 waves ----
        for (int job = wave; job < 48; job += 16) {
          int isg = (job >= 16);
          int nt  = isg ? (job - 16) : job;
          const __bf16* w3p = isg ? s_gw3 : s_fw3;
          v8f c = {};
#pragma unroll
          for (int k0 = 0; k0 < 64; k0 += 32) {
            v16bf a  = load_frag(s_h2 + lm * P128 + (isg ? 64 : 0), k0, hi);
            v16bf bm = load_frag(w3p + (nt * 16 + lm) * P64,        k0, hi);
            c = wmma_bf16(a, bm, c);
          }
          int n = nt * 16 + lm;
          float bias = isg ? s_gb3[n] : s_fb3[n];
          float* outp = isg ? s_gout : s_fout;
          int ldo = isg ? 512 : 256;
#pragma unroll
          for (int v = 0; v < 8; ++v)
            outp[(v + 8 * hi) * ldo + n] = c[v] + bias;
        }
        __syncthreads();
        // ---- combine: k = f + g.u ; acc += w*k ; next eval point ----
        {
          float wv = (st == 1 || st == 2) ? 2.f : 1.f;
          float cv = (st == 2) ? hh : 0.5f * hh;
          for (int q = 0; q < 8; ++q) {
            int e = tid * 8 + q; int r = e >> 8, l = e & 255;
            float kv = s_fout[e] + s_gout[r * 512 + 2 * l]     * s_uk[r * 2]
                                 + s_gout[r * 512 + 2 * l + 1] * s_uk[r * 2 + 1];
            s_acc[e] += wv * kv;
            if (st < 3) s_xbf[r * P256 + l] = (__bf16)(s_x0[e] + cv * kv);
          }
        }
        __syncthreads();
      }
      for (int q = 0; q < 8; ++q) {
        int e = tid * 8 + q;
        s_x0[e] += (hh / 6.f) * s_acc[e];
      }
    }
    for (int q = 0; q < 8; ++q) {
      int e = tid * 8 + q; int r = e >> 8, l = e & 255;
      tides[((size_t)(b0 + r) * TT + k) * LATENT + l] = s_x0[e];
    }
  }
}

// ============================================================================
extern "C" void kernel_launch(void* const* d_in, const int* in_sizes, int n_in,
                              void* d_out, int out_size, void* d_ws, size_t ws_size,
                              hipStream_t stream) {
  const float* gi     = (const float*)d_in[0];
  const float* gu     = (const float*)d_in[1];
  const float* phi_w1 = (const float*)d_in[2];
  const float* phi_b1 = (const float*)d_in[3];
  const float* phi_w2 = (const float*)d_in[4];
  const float* phi_b2 = (const float*)d_in[5];
  const float* rho_w  = (const float*)d_in[6];
  const float* rho_b  = (const float*)d_in[7];
  const float* f_w1   = (const float*)d_in[8];
  const float* f_b1   = (const float*)d_in[9];
  const float* f_w2   = (const float*)d_in[10];
  const float* f_b2   = (const float*)d_in[11];
  const float* f_w3   = (const float*)d_in[12];
  const float* f_b3   = (const float*)d_in[13];
  const float* g_w1   = (const float*)d_in[14];
  const float* g_b1   = (const float*)d_in[15];
  const float* g_w2   = (const float*)d_in[16];
  const float* g_b2   = (const float*)d_in[17];
  const float* g_w3   = (const float*)d_in[18];
  const float* g_b3   = (const float*)d_in[19];

  float* xs    = (float*)d_out;                       // [B,T,L]
  float* tides = xs + (size_t)BB * TT * LATENT;       // [B,T,L]

  const size_t ENC_SMEM =
      (size_t)(64 + 64 + 64 + 256) * 4 +
      (size_t)(512 * P64 + 512 * P64 + 64 * P64 + 256 * P64 + 16 * P64) * 2;
  const size_t SCAN_SMEM =
      (size_t)(4096 * 3 + 8192 + 64 * 4 + 256 + 512 + 32) * 4 +
      (size_t)(128 * P256 + 64 * P64 * 2 + 256 * P64 + 512 * P64 +
               16 * P256 + 16 * P128 * 2) * 2;

  (void)hipFuncSetAttribute((const void*)encoder_kernel,
                            hipFuncAttributeMaxDynamicSharedMemorySize,
                            (int)ENC_SMEM);
  (void)hipFuncSetAttribute((const void*)scan_kernel,
                            hipFuncAttributeMaxDynamicSharedMemorySize,
                            (int)SCAN_SMEM);

  dim3 egrid((TT + 15) / 16, BB);
  encoder_kernel<<<egrid, 256, ENC_SMEM, stream>>>(
      gi, phi_w1, phi_b1, phi_w2, phi_b2, rho_w, rho_b, xs);

  scan_kernel<<<BB / 16, 512, SCAN_SMEM, stream>>>(
      gu, f_w1, f_b1, f_w2, f_b2, f_w3, f_b3,
      g_w1, g_b1, g_w2, g_b2, g_w3, g_b3, xs, tides);
}
